// MultiScaleFusionBlock_28329604285170
// MI455X (gfx1250) — compile-verified, matrix operand order checked by reading, not run on hardware
//
#include <hip/hip_runtime.h>
#include <hip/hip_bf16.h>

// ---------------- types for CDNA5 WMMA / TDM ----------------
typedef __attribute__((ext_vector_type(16))) __bf16 v16bf;
typedef __attribute__((ext_vector_type(8)))  __bf16 v8bf;
typedef __attribute__((ext_vector_type(8)))  float  v8f;
typedef __attribute__((ext_vector_type(4)))  unsigned v4u;
typedef __attribute__((ext_vector_type(8)))  unsigned v8u;

// ---------------- problem constants ----------------
constexpr int BATCH  = 2;
constexpr int CSUM   = 800;
constexpr int DIN    = 1600;
constexpr int NXZ    = 3200;     // 2*DIN
constexpr int DSTATE = 16;
constexpr int LTOT   = 4096;     // 16*16*16
constexpr int MROWS  = BATCH * LTOT;   // 8192
constexpr int DTRANK = 50;
constexpr int DTPADK = 64;       // dt K padded for WMMA
constexpr int XPN    = 96;       // x_proj N padded (82 -> 96)
constexpr int CHUNK  = 256;      // scan TDM chunk (rows per tensor tile)
constexpr int NCHUNK = LTOT / CHUNK;

// ============================================================
// helpers
// ============================================================
__device__ __forceinline__ float tri3(const float* p, int S,
                                      long sx, long sy, long sz,
                                      float fz, float fy, float fx) {
  int z0 = (int)fz, y0 = (int)fy, x0 = (int)fx;
  int z1 = z0 + 1; if (z1 > S - 1) z1 = S - 1;
  int y1 = y0 + 1; if (y1 > S - 1) y1 = S - 1;
  int x1 = x0 + 1; if (x1 > S - 1) x1 = S - 1;
  float wz = fz - (float)z0, wy = fy - (float)y0, wx = fx - (float)x0;
  auto at = [&](int z, int y, int x) -> float {
    return p[(long)z * sz + (long)y * sy + (long)x * sx];
  };
  float c00 = at(z0,y0,x0) * (1.f-wx) + at(z0,y0,x1) * wx;
  float c01 = at(z0,y1,x0) * (1.f-wx) + at(z0,y1,x1) * wx;
  float c10 = at(z1,y0,x0) * (1.f-wx) + at(z1,y0,x1) * wx;
  float c11 = at(z1,y1,x0) * (1.f-wx) + at(z1,y1,x1) * wx;
  float c0  = c00 * (1.f-wy) + c01 * wy;
  float c1  = c10 * (1.f-wy) + c11 * wy;
  return c0 * (1.f-wz) + c1 * wz;
}

// ============================================================
// weight prep
// ============================================================
__global__ void k_convert_bf16(const float* __restrict__ src, __bf16* __restrict__ dst, long n) {
  long i = (long)blockIdx.x * blockDim.x + threadIdx.x;
  if (i < n) dst[i] = (__bf16)src[i];
}

__global__ void k_pad_xproj(const float* __restrict__ src, __bf16* __restrict__ dst) {
  long i = (long)blockIdx.x * blockDim.x + threadIdx.x;
  long n = (long)XPN * DIN;
  if (i >= n) return;
  int row = (int)(i / DIN);
  dst[i] = (row < 82) ? (__bf16)src[i] : (__bf16)0.f;
}

__global__ void k_pad_dtw(const float* __restrict__ src, __bf16* __restrict__ dst) {
  long i = (long)blockIdx.x * blockDim.x + threadIdx.x;
  long n = (long)DIN * DTPADK;
  if (i >= n) return;
  int col = (int)(i & (DTPADK - 1));
  long row = i >> 6;
  dst[i] = (col < DTRANK) ? (__bf16)src[row * DTRANK + col] : (__bf16)0.f;
}

// ============================================================
// multi-scale fusion -> concat (B*L, 800) f32, channel-last
// ============================================================
__global__ void k_fuse(const float* __restrict__ s0, const float* __restrict__ s1,
                       const float* __restrict__ s2, const float* __restrict__ s3,
                       const float* __restrict__ s4, float* __restrict__ concat) {
  long idx = (long)blockIdx.x * blockDim.x + threadIdx.x;
  long n = (long)MROWS * CSUM;
  if (idx >= n) return;
  int c = (int)(idx % CSUM);
  long bl = idx / CSUM;
  int b = (int)(bl / LTOT);
  int l = (int)(bl % LTOT);
  int z = l >> 8, y = (l >> 4) & 15, x = l & 15;
  float v;
  if (c < 32) {
    const float* p = s0 + ((long)b * 32 + c) * (64L*64*64);
    float acc = 0.f;
    for (int dz = 0; dz < 4; dz++)
      for (int dy = 0; dy < 4; dy++)
        for (int dx = 0; dx < 4; dx++)
          acc += p[((long)(z*4+dz)*64 + (y*4+dy))*64 + (x*4+dx)];
    v = acc * (1.f/64.f);
  } else if (c < 96) {
    int cc = c - 32;
    const float* p = s1 + ((long)b * 64 + cc) * (32L*32*32);
    float acc = 0.f;
    for (int dz = 0; dz < 2; dz++)
      for (int dy = 0; dy < 2; dy++)
        for (int dx = 0; dx < 2; dx++)
          acc += p[((long)(z*2+dz)*32 + (y*2+dy))*32 + (x*2+dx)];
    v = acc * (1.f/8.f);
  } else if (c < 224) {
    int cc = c - 96;
    v = s2[(((long)b * 128 + cc) * 16 + z) * 256 + y * 16 + x];
  } else if (c < 480) {
    int cc = c - 224;
    const float* p = s3 + ((long)b * 256 + cc) * 512;
    const float sc = 7.f / 15.f;
    v = tri3(p, 8, 1, 8, 64, z*sc, y*sc, x*sc);
  } else {
    int cc = c - 480;
    const float* p = s4 + ((long)b * 320 + cc) * 64;
    v = tri3(p, 4, 1, 4, 16, z*0.2f, y*0.2f, x*0.2f);
  }
  concat[idx] = v;
}

// ============================================================
// layernorm over C=800 per row -> bf16 activations
// ============================================================
__global__ void k_layernorm(const float* __restrict__ concat,
                            const float* __restrict__ w, const float* __restrict__ bia,
                            __bf16* __restrict__ ta) {
  int row = blockIdx.x;
  const float* xr = concat + (long)row * CSUM;
  __shared__ float red[256];
  int tid = threadIdx.x;
  float s = 0.f;
  for (int i = tid; i < CSUM; i += 256) s += xr[i];
  red[tid] = s; __syncthreads();
  for (int st = 128; st > 0; st >>= 1) { if (tid < st) red[tid] += red[tid + st]; __syncthreads(); }
  float mean = red[0] * (1.f / CSUM);
  __syncthreads();
  float v = 0.f;
  for (int i = tid; i < CSUM; i += 256) { float d = xr[i] - mean; v += d * d; }
  red[tid] = v; __syncthreads();
  for (int st = 128; st > 0; st >>= 1) { if (tid < st) red[tid] += red[tid + st]; __syncthreads(); }
  float rstd = rsqrtf(red[0] * (1.f / CSUM) + 1e-5f);
  for (int i = tid; i < CSUM; i += 256)
    ta[(long)row * CSUM + i] = (__bf16)((xr[i] - mean) * rstd * w[i] + bia[i]);
}

// ============================================================
// WMMA bf16 GEMM:  D[M,N] = A[M,K] * W[N,K]^T  (f32 accum)
// register-tiled: one wave -> (16*MT) x (16*NT) tile, K step 32
// mode 0: plain store; mode 1: softplus(acc + bias[n])
// ============================================================
__device__ __forceinline__ v16bf load_frag(const __bf16* base, int ld, int k, int lane) {
  int r = lane & 15, h = lane >> 4;
  const __bf16* p = base + (long)r * ld + k + h * 8;
  v8bf lo = *reinterpret_cast<const v8bf*>(p);
  v8bf hi = *reinterpret_cast<const v8bf*>(p + 16);
  v16bf o;
#pragma unroll
  for (int i = 0; i < 8; i++) { o[i] = lo[i]; o[i + 8] = hi[i]; }
  return o;
}

template <int MT, int NT>
__global__ void k_gemm_t(const __bf16* __restrict__ A, const __bf16* __restrict__ W,
                         float* __restrict__ D, int Mm, int Nn, int Kk,
                         const float* __restrict__ bias, int mode) {
  int wave = threadIdx.x >> 5;
  int lane = threadIdx.x & 31;
  long tile = (long)blockIdx.x * 8 + wave;
  int tn = Nn / (16 * NT);
  long total = (long)(Mm / (16 * MT)) * tn;
  if (tile >= total) return;
  int ti = (int)(tile / tn), tj = (int)(tile % tn);
  int m0 = ti * 16 * MT, n0 = tj * 16 * NT;

  v8f acc[MT][NT] = {};
  for (int k = 0; k < Kk; k += 32) {
    if (k + 32 < Kk)   // gfx1250 global_prefetch_b8 of next K slice
      __builtin_prefetch((const void*)(A + (long)(m0 + (lane & 15)) * Kk + k + 32), 0, 1);
    v16bf af[MT], bf[NT];
#pragma unroll
    for (int i = 0; i < MT; i++) af[i] = load_frag(A + (long)(m0 + 16 * i) * Kk, Kk, k, lane);
#pragma unroll
    for (int j = 0; j < NT; j++) bf[j] = load_frag(W + (long)(n0 + 16 * j) * Kk, Kk, k, lane);
#pragma unroll
    for (int i = 0; i < MT; i++)
#pragma unroll
      for (int j = 0; j < NT; j++)
        acc[i][j] = __builtin_amdgcn_wmma_f32_16x16x32_bf16(
            false, af[i], false, bf[j], (short)0, acc[i][j], false, false);
  }
  int r = lane & 15, h = lane >> 4;
#pragma unroll
  for (int i = 0; i < MT; i++) {
#pragma unroll
    for (int j = 0; j < NT; j++) {
#pragma unroll
      for (int v = 0; v < 8; v++) {
        int m = m0 + 16 * i + v + h * 8;
        int cn = n0 + 16 * j + r;
        float x = acc[i][j][v];
        if (mode == 1) {
          x += bias[cn];
          x = (x > 20.f) ? x : log1pf(__expf(x));
        }
        D[(long)m * Nn + cn] = x;
      }
    }
  }
}

// ============================================================
// causal depthwise conv (k=4) + SiLU over x half of xz
// ============================================================
__global__ void k_conv_silu(const float* __restrict__ xz, const float* __restrict__ cw,
                            const float* __restrict__ cb, __bf16* __restrict__ xconv) {
  long idx = (long)blockIdx.x * blockDim.x + threadIdx.x;
  long n = (long)MROWS * DIN;
  if (idx >= n) return;
  int d = (int)(idx % DIN);
  long bl = idx / DIN;
  int b = (int)(bl / LTOT);
  int l = (int)(bl % LTOT);
  float acc = cb[d];
#pragma unroll
  for (int k = 0; k < 4; k++) {
    int ls = l + k - 3;
    if (ls >= 0) acc += cw[d * 4 + k] * xz[((long)b * LTOT + ls) * NXZ + d];
  }
  float s = acc / (1.f + __expf(-acc));
  xconv[idx] = (__bf16)s;
}

__global__ void k_dtpad(const float* __restrict__ dbl, __bf16* __restrict__ dtp) {
  long idx = (long)blockIdx.x * blockDim.x + threadIdx.x;
  long n = (long)MROWS * DTPADK;
  if (idx >= n) return;
  int c = (int)(idx & (DTPADK - 1));
  long r = idx >> 6;
  dtp[idx] = (c < DTRANK) ? (__bf16)dbl[r * XPN + c] : (__bf16)0.f;
}

// ============================================================
// selective scan: serial over L, parallel over (b, d)
// per-step B/C vectors staged through LDS by the Tensor Data Mover
// (tensor_load_to_lds, double-buffered 32x256 f32 tiles, TENSORcnt sync)
// ============================================================
__global__ void k_scan(const float* __restrict__ delta, const float* __restrict__ dbl,
                       const __bf16* __restrict__ xconv, const float* __restrict__ xz,
                       const float* __restrict__ A_log, const float* __restrict__ Dp,
                       __bf16* __restrict__ yout) {
  int b = blockIdx.y;
  int d = blockIdx.x * 64 + threadIdx.x;
  float Ar[DSTATE], h[DSTATE];
#pragma unroll
  for (int nn = 0; nn < DSTATE; nn++) { Ar[nn] = -__expf(A_log[d * DSTATE + nn]); h[nn] = 0.f; }
  float dpar = Dp[d];

  __shared__ float sbc[2][CHUNK * 32];   // [buf][row*32 + (B:0..15 | C:16..31)]
  long rowbase = (long)b * LTOT;

  // Build D# (groups 0,1) per CDNA5 ISA and issue TDM.
  auto issue = [&](int chunk, int buf) {
    unsigned long long ga =
        (unsigned long long)(const void*)(dbl + (rowbase + (long)chunk * CHUNK) * XPN + DTRANK);
    unsigned lds = (unsigned)(unsigned long long)(size_t)&sbc[buf][0];
    v4u g0;
    g0[0] = 1u;                                   // count=1, user desc
    g0[1] = lds;                                  // lds_addr
    g0[2] = (unsigned)ga;                         // global_addr[31:0]
    g0[3] = (unsigned)((ga >> 32) & 0x01FFFFFFu)  // global_addr[56:32]
            | (2u << 30);                         // type=2 (image)
    v8u g1;
    g1[0] = (2u << 16);                           // wg_mask=0, data_size=4B
    g1[1] = (32u << 16);                          // tensor_dim0=32 (bits 63:48)
    g1[2] = ((unsigned)CHUNK << 16);              // dim0 hi=0, tensor_dim1=CHUNK (bits 95:80)
    g1[3] = (32u << 16);                          // dim1 hi=0, tile_dim0=32 (bits 127:112)
    g1[4] = (unsigned)CHUNK;                      // tile_dim1=CHUNK, tile_dim2=0
    g1[5] = (unsigned)XPN;                        // tensor_dim0_stride=96 (bits 191:160)
    g1[6] = 0u;                                   // stride0 hi, stride1 lo
    g1[7] = 0u;                                   // stride1 hi
    asm volatile("tensor_load_to_lds %0, %1" :: "s"(g0), "s"(g1) : "memory");
  };

  if (threadIdx.x < 32) issue(0, 0);              // wave 0 drives the TDM

  for (int c = 0; c < NCHUNK; ++c) {
    if (threadIdx.x < 32) __builtin_amdgcn_s_wait_tensorcnt(0);
    __syncthreads();                              // chunk c visible to both waves
    if (c + 1 < NCHUNK && threadIdx.x < 32) issue(c + 1, (c + 1) & 1);
    const float* bc = sbc[c & 1];
    for (int t = 0; t < CHUNK; ++t) {
      long row = rowbase + (long)c * CHUNK + t;
      float dl = delta[row * DIN + d];
      float u = (float)xconv[row * DIN + d];
      float du = dl * u;
      float y = 0.f;
      const float* pb = bc + t * 32;
#pragma unroll
      for (int nn = 0; nn < DSTATE; nn++) {
        h[nn] = h[nn] * __expf(dl * Ar[nn]) + du * pb[nn];
        y += h[nn] * pb[DSTATE + nn];
      }
      float z = xz[row * NXZ + DIN + d];
      float yf = (y + u * dpar) * (z / (1.f + __expf(-z)));
      yout[row * DIN + d] = (__bf16)yf;
    }
    __syncthreads();                              // done reading buf before TDM reuses it
  }
}

// ============================================================
// un-fuse: inverse resample each channel slice + skip add
// ============================================================
__global__ void k_un_up(const float* __restrict__ mout, int coff, int C,
                        const float* __restrict__ skip, float* __restrict__ out, int S) {
  long idx = (long)blockIdx.x * blockDim.x + threadIdx.x;
  long total = (long)BATCH * C * S * S * S;
  if (idx >= total) return;
  int x = (int)(idx % S); long t = idx / S;
  int y = (int)(t % S); t /= S;
  int z = (int)(t % S); t /= S;
  int c = (int)(t % C);
  int b = (int)(t / C);
  float sc = 15.f / (float)(S - 1);
  const float* p = mout + (long)b * LTOT * CSUM + coff + c;
  float v = tri3(p, 16, CSUM, 16L * CSUM, 256L * CSUM, z * sc, y * sc, x * sc);
  out[idx] = v + skip[idx];
}

__global__ void k_un_copy(const float* __restrict__ mout, int coff,
                          const float* __restrict__ skip, float* __restrict__ out) {
  long idx = (long)blockIdx.x * blockDim.x + threadIdx.x;
  long total = (long)BATCH * 128 * LTOT;
  if (idx >= total) return;
  int l = (int)(idx % LTOT); long t = idx / LTOT;
  int c = (int)(t % 128);
  int b = (int)(t / 128);
  float v = mout[((long)b * LTOT + l) * CSUM + coff + c];
  out[idx] = v + skip[idx];
}

__global__ void k_un_pool(const float* __restrict__ mout, int coff, int C,
                          const float* __restrict__ skip, float* __restrict__ out,
                          int S, int s) {
  long idx = (long)blockIdx.x * blockDim.x + threadIdx.x;
  long total = (long)BATCH * C * S * S * S;
  if (idx >= total) return;
  int x = (int)(idx % S); long t = idx / S;
  int y = (int)(t % S); t /= S;
  int z = (int)(t % S); t /= S;
  int c = (int)(t % C);
  int b = (int)(t / C);
  float acc = 0.f;
  for (int dz = 0; dz < s; dz++)
    for (int dy = 0; dy < s; dy++)
      for (int dx = 0; dx < s; dx++) {
        int l = ((z * s + dz) * 16 + (y * s + dy)) * 16 + (x * s + dx);
        acc += mout[((long)b * LTOT + l) * CSUM + coff + c];
      }
  out[idx] = acc / (float)(s * s * s) + skip[idx];
}

// ============================================================
// launcher
// ============================================================
static inline int nblk(long n, int bs) { return (int)((n + bs - 1) / bs); }

extern "C" void kernel_launch(void* const* d_in, const int* in_sizes, int n_in,
                              void* d_out, int out_size, void* d_ws, size_t ws_size,
                              hipStream_t stream) {
  const float* s0 = (const float*)d_in[0];
  const float* s1 = (const float*)d_in[1];
  const float* s2 = (const float*)d_in[2];
  const float* s3 = (const float*)d_in[3];
  const float* s4 = (const float*)d_in[4];
  const float* ln_w  = (const float*)d_in[5];
  const float* ln_b  = (const float*)d_in[6];
  const float* inpw  = (const float*)d_in[7];
  const float* convw = (const float*)d_in[8];
  const float* convb = (const float*)d_in[9];
  const float* xprojw  = (const float*)d_in[10];
  const float* dtprojw = (const float*)d_in[11];
  const float* dtprojb = (const float*)d_in[12];
  const float* alog    = (const float*)d_in[13];
  const float* dparam  = (const float*)d_in[14];
  const float* outpw   = (const float*)d_in[15];

  char* ws = (char*)d_ws;
  size_t off = 0;
  auto alloc = [&](size_t bytes) -> void* {
    void* p = ws + off;
    off = (off + bytes + 255) & ~(size_t)255;
    return p;
  };
  __bf16* wIn   = (__bf16*)alloc((size_t)NXZ * CSUM * 2);
  __bf16* wXp   = (__bf16*)alloc((size_t)XPN * DIN * 2);
  __bf16* wDt   = (__bf16*)alloc((size_t)DIN * DTPADK * 2);
  __bf16* wOut  = (__bf16*)alloc((size_t)CSUM * DIN * 2);
  float*  concat= (float*)alloc((size_t)MROWS * CSUM * 4);
  __bf16* ta    = (__bf16*)alloc((size_t)MROWS * CSUM * 2);
  float*  xz    = (float*)alloc((size_t)MROWS * NXZ * 4);
  __bf16* xconv = (__bf16*)alloc((size_t)MROWS * DIN * 2);
  float*  dbl   = (float*)alloc((size_t)MROWS * XPN * 4);
  __bf16* dtp   = (__bf16*)alloc((size_t)MROWS * DTPADK * 2);
  float*  delta = (float*)alloc((size_t)MROWS * DIN * 4);
  __bf16* yout  = (__bf16*)alloc((size_t)MROWS * DIN * 2);
  float*  mout  = (float*)alloc((size_t)MROWS * CSUM * 4);

  // weight prep (bf16 + padding)
  k_convert_bf16<<<nblk((long)NXZ * CSUM, 256), 256, 0, stream>>>(inpw, wIn, (long)NXZ * CSUM);
  k_convert_bf16<<<nblk((long)CSUM * DIN, 256), 256, 0, stream>>>(outpw, wOut, (long)CSUM * DIN);
  k_pad_xproj<<<nblk((long)XPN * DIN, 256), 256, 0, stream>>>(xprojw, wXp);
  k_pad_dtw<<<nblk((long)DIN * DTPADK, 256), 256, 0, stream>>>(dtprojw, wDt);

  // fuse -> concat -> layernorm -> bf16 activations
  k_fuse<<<nblk((long)MROWS * CSUM, 256), 256, 0, stream>>>(s0, s1, s2, s3, s4, concat);
  k_layernorm<<<MROWS, 256, 0, stream>>>(concat, ln_w, ln_b, ta);

  // in_proj GEMM: (8192 x 800) x (3200 x 800)^T  -- 32x64 wave tiles
  {
    long tiles = (long)(MROWS / 32) * (NXZ / 64);
    k_gemm_t<2,4><<<nblk(tiles, 8), 256, 0, stream>>>(ta, wIn, xz, MROWS, NXZ, CSUM, nullptr, 0);
  }

  // conv + silu
  k_conv_silu<<<nblk((long)MROWS * DIN, 256), 256, 0, stream>>>(xz, convw, convb, xconv);

  // x_proj GEMM: (8192 x 1600) x (96 x 1600)^T  -- 32x32 wave tiles
  {
    long tiles = (long)(MROWS / 32) * (XPN / 32);
    k_gemm_t<2,2><<<nblk(tiles, 8), 256, 0, stream>>>(xconv, wXp, dbl, MROWS, XPN, DIN, nullptr, 0);
  }

  // dt pad + dt_proj GEMM with fused softplus(acc + bias) -- 32x64 wave tiles
  k_dtpad<<<nblk((long)MROWS * DTPADK, 256), 256, 0, stream>>>(dbl, dtp);
  {
    long tiles = (long)(MROWS / 32) * (DIN / 64);
    k_gemm_t<2,4><<<nblk(tiles, 8), 256, 0, stream>>>(dtp, wDt, delta, MROWS, DIN, DTPADK, dtprojb, 1);
  }

  // selective scan (TDM-staged B/C, fused D-skip + silu(z) gating)
  k_scan<<<dim3(DIN / 64, BATCH), 64, 0, stream>>>(delta, dbl, xconv, xz, alog, dparam, yout);

  // out_proj GEMM: (8192 x 1600) x (800 x 1600)^T  -- 32x32 wave tiles
  {
    long tiles = (long)(MROWS / 32) * (CSUM / 32);
    k_gemm_t<2,2><<<nblk(tiles, 8), 256, 0, stream>>>(yout, wOut, mout, MROWS, CSUM, DIN, nullptr, 0);
  }

  // un-fuse + skip adds, outputs concatenated flat
  float* out = (float*)d_out;
  const long o0 = 0;
  const long o1 = o0 + (long)BATCH * 32 * 64L * 64 * 64;   // 16777216
  const long o2 = o1 + (long)BATCH * 64 * 32L * 32 * 32;   // 20971520
  const long o3 = o2 + (long)BATCH * 128 * 16L * 16 * 16;  // 22020096
  const long o4 = o3 + (long)BATCH * 256 * 8L * 8 * 8;     // 22282240

  k_un_up  <<<nblk((long)BATCH * 32 * 64L*64*64, 256), 256, 0, stream>>>(mout, 0,   32,  s0, out + o0, 64);
  k_un_up  <<<nblk((long)BATCH * 64 * 32L*32*32, 256), 256, 0, stream>>>(mout, 32,  64,  s1, out + o1, 32);
  k_un_copy<<<nblk((long)BATCH * 128 * (long)LTOT, 256), 256, 0, stream>>>(mout, 96, s2, out + o2);
  k_un_pool<<<nblk((long)BATCH * 256 * 8L*8*8, 256), 256, 0, stream>>>(mout, 224, 256, s3, out + o3, 8, 2);
  k_un_pool<<<nblk((long)BATCH * 320 * 4L*4*4, 256), 256, 0, stream>>>(mout, 480, 320, s4, out + o4, 4, 4);
}
